// WindowAttention_79594333929518
// MI455X (gfx1250) — matrix operand order repaired and checked by reading, not compiled, not run on hardware
//
#include <hip/hip_runtime.h>

// ---------------------------------------------------------------------------
// Swin shifted-window attention forward for MI455X (gfx1250, wave32, WMMA).
// bf16 tensor-core path: v_wmma_f32_16x16x32_bf16 for all three matmul stages.
// B operands staged into LDS in B-fragment-major (column-major) order with
// K-pairs packed as dwords: ds_store_b32/b128 staging, ds_load_b128 fragments,
// v_cvt_pk_bf16_f32 conversions via vector converts.
// ---------------------------------------------------------------------------

#define DISP 4
#define TSZ  (32*8*64*64*32)   // elements per q/k/v tensor = 33,554,432

typedef __attribute__((ext_vector_type(16))) unsigned short v16u;
typedef __attribute__((ext_vector_type(8)))  unsigned short v8u;
typedef __attribute__((ext_vector_type(16))) __bf16         v16bf;
typedef __attribute__((ext_vector_type(2)))  __bf16         v2bf;
typedef __attribute__((ext_vector_type(2)))  float          v2f;
typedef __attribute__((ext_vector_type(8)))  float          v8f;

__device__ __forceinline__ unsigned short f2bf_u(float f) {
  __bf16 h = (__bf16)f;                       // native RTE convert
  return __builtin_bit_cast(unsigned short, h);
}

// pack two floats as adjacent bf16 -> one v_cvt_pk_bf16_f32 with (a,b) operands
__device__ __forceinline__ unsigned int pkbf2(float a, float b) {
  v2f f = {a, b};
  v2bf t = __builtin_convertvector(f, v2bf);
  return __builtin_bit_cast(unsigned int, t);
}

__device__ __forceinline__ unsigned int pk2(unsigned short a, unsigned short b) {
  return (unsigned int)a | ((unsigned int)b << 16);
}

__device__ __forceinline__ v16u cat8(v8u lo, v8u hi) {
  union { struct { v8u a, b; } s; v16u v; } u;
  u.s.a = lo; u.s.b = hi;
  return u.v;
}

__device__ __forceinline__ v8f wmma_bf16(v16u a, v16u b, v8f c) {
  return __builtin_amdgcn_wmma_f32_16x16x32_bf16(
      false, __builtin_bit_cast(v16bf, a),
      false, __builtin_bit_cast(v16bf, b),
      (short)0, c, false, false);
}

// ---------------------------------------------------------------------------
// Kernel 1: qkv = roll(x,-4,-4) @ w_qkv, scale folded into q, written bf16 in
// window-major layout qkv_ws[which][b][head][win][tok][d].
// Block: 256 thr (8 waves as 2x4), tile 64(M) x 128(N), K-chunks of 32.
// LDS: As row-major 64x32; Bsr column-major 128(n) x 32(k).
// ---------------------------------------------------------------------------
__global__ __launch_bounds__(256)
void qkv_kernel(const float* __restrict__ x, const float* __restrict__ w_qkv,
                unsigned short* __restrict__ qkv_ws) {
  __shared__ __align__(32) unsigned short As[64*32];
  __shared__ __align__(32) unsigned short Bsr[128*32];   // [n][k]
  unsigned int* Bsr32 = (unsigned int*)Bsr;              // [n][kpair]
  const int tid  = threadIdx.x;
  const int mt   = blockIdx.x;        // 0..2047
  const int n0   = blockIdx.y * 128;  // 0..640
  const int lane = tid & 31, wave = tid >> 5;
  const int wm = wave >> 2, wn = wave & 3;
  const int halfl = lane >> 4, rowl = lane & 15;

  v8f acc[2][2] = {};

  // A staging: thread -> (row, 8 cols); cyclic shift fused into source address
  const int sr = tid >> 2, sc = (tid & 3) * 8;
  const int m_s = mt * 64 + sr;
  const int bb = m_s >> 12, hh = (m_s >> 6) & 63, ww = m_s & 63;
  const int hs = (hh + DISP) & 63, wsr = (ww + DISP) & 63;
  const float* xsrc = x + (size_t)((bb << 12) + (hs << 6) + wsr) * 256 + sc;
  // B staging: thread owns K-pair (2*bkp, 2*bkp+1) x 8 n-values
  const int bkp = tid >> 4, bn0 = (tid & 15) * 8;
  const float* bsrc0 = w_qkv + (size_t)(2 * bkp) * 768 + n0 + bn0;

  for (int kc = 0; kc < 8; ++kc) {
    const int k0 = kc * 32;
    if (kc < 7) {  // prefetch next K-chunk -> global_prefetch_b8
      __builtin_prefetch(xsrc + k0 + 32, 0, 3);
      __builtin_prefetch(bsrc0 + (size_t)(k0 + 32) * 768, 0, 3);
    }
    { // stage A tile (f32 -> bf16), row-major; packs into one b128 store
      const float4* p = (const float4*)(xsrc + k0);
      float4 f0 = p[0], f1 = p[1];
      unsigned int* d = (unsigned int*)&As[sr * 32 + sc];
      d[0] = pkbf2(f0.x, f0.y); d[1] = pkbf2(f0.z, f0.w);
      d[2] = pkbf2(f1.x, f1.y); d[3] = pkbf2(f1.z, f1.w);
    }
    { // stage B transposed with packed K-pairs: Bsr32[n][kp]
      const float4* p0 = (const float4*)(bsrc0 + (size_t)k0 * 768);
      const float4* p1 = (const float4*)(bsrc0 + (size_t)k0 * 768 + 768);
      float4 a0 = p0[0], a1 = p0[1];     // row k = k0+2*bkp
      float4 b0 = p1[0], b1 = p1[1];     // row k = k0+2*bkp+1
      Bsr32[(bn0 + 0) * 16 + bkp] = pkbf2(a0.x, b0.x);
      Bsr32[(bn0 + 1) * 16 + bkp] = pkbf2(a0.y, b0.y);
      Bsr32[(bn0 + 2) * 16 + bkp] = pkbf2(a0.z, b0.z);
      Bsr32[(bn0 + 3) * 16 + bkp] = pkbf2(a0.w, b0.w);
      Bsr32[(bn0 + 4) * 16 + bkp] = pkbf2(a1.x, b1.x);
      Bsr32[(bn0 + 5) * 16 + bkp] = pkbf2(a1.y, b1.y);
      Bsr32[(bn0 + 6) * 16 + bkp] = pkbf2(a1.z, b1.z);
      Bsr32[(bn0 + 7) * 16 + bkp] = pkbf2(a1.w, b1.w);
    }
    __syncthreads();
    v16u afrag[2];
    #pragma unroll
    for (int mi = 0; mi < 2; ++mi) {
      const unsigned short* ap = &As[(wm*32 + mi*16 + rowl) * 32];
      afrag[mi] = *(const v16u*)(ap + halfl * 16);
    }
    #pragma unroll
    for (int nj = 0; nj < 2; ++nj) {
      const int bc = wn*32 + nj*16 + rowl;
      const v16u bu = *(const v16u*)(&Bsr[bc * 32 + halfl * 16]);  // contiguous
      #pragma unroll
      for (int mi = 0; mi < 2; ++mi)
        acc[mi][nj] = wmma_bf16(afrag[mi], bu, acc[mi][nj]);
    }
    __syncthreads();
  }
  // Epilogue: per-lane row base is 8-aligned -> the 8 rows are consecutive
  // tokens in one window; store index is affine: idx0 + r*32.
  const float scale = 0.1767766952966369f;  // 1/sqrt(32), folded into q
  #pragma unroll
  for (int mi = 0; mi < 2; ++mi) {
    const int gm0 = mt*64 + wm*32 + mi*16 + halfl*8;   // multiple of 8
    const int b2 = gm0 >> 12, h2 = (gm0 >> 6) & 63, w2 = gm0 & 63;
    const int win = ((h2 >> 3) << 3) + (w2 >> 3);
    const int tok0 = (h2 & 7) << 3;                    // w2 & 7 == 0
    #pragma unroll
    for (int nj = 0; nj < 2; ++nj) {
      const int gn = n0 + wn*32 + nj*16 + rowl;
      const int which = gn >> 8, head = (gn >> 5) & 7, d = gn & 31;
      const float s = (gn < 256) ? scale : 1.0f;
      size_t idx0 = (size_t)which * TSZ +
                    ((((size_t)(b2*8 + head) * 64 + win) * 64 + tok0) * 32 + d);
      #pragma unroll
      for (int r = 0; r < 8; ++r)
        qkv_ws[idx0 + (size_t)r * 32] = f2bf_u(acc[mi][nj][r] * s);
    }
  }
}

// ---------------------------------------------------------------------------
// Kernel 2: per-(b,head,window) attention. 4 waves / block; wave w owns score
// rows [16w,16w+16). S = qK^T (WMMA), bias+masks+softmax in LDS, O = P V (WMMA).
// V staged transposed (Vt[d][tok], tok-pairs packed) so P@V B-fragments load
// contiguously.
// ---------------------------------------------------------------------------
__global__ __launch_bounds__(128)
void attn_kernel(const unsigned short* __restrict__ qkv,
                 const float* __restrict__ pos_emb,
                 unsigned short* __restrict__ o) {
  __shared__ float Sf[64][65];
  __shared__ __align__(32) unsigned short Pb[64][64];
  __shared__ __align__(32) unsigned short Vt[32*64];   // [d][tok]
  __shared__ float pe[225];
  unsigned int* Vt32 = (unsigned int*)Vt;              // [d][tokpair]

  const int tid = threadIdx.x;
  const int id = blockIdx.x;
  const int win = id & 63, head = (id >> 6) & 7, b = id >> 9;
  const size_t base = (size_t)((b*8 + head) * 64 + win) * (64 * 32);
  const unsigned short* q = qkv + base;
  const unsigned short* k = qkv + (size_t)TSZ + base;
  const unsigned short* v = qkv + (size_t)2 * TSZ + base;

  for (int i = tid; i < 225; i += 128) pe[i] = pos_emb[i];
  { // stage V transposed with packed tok-pairs: Vt32[d][tokp]
    const int tokp = tid >> 2, d0v = (tid & 3) * 8;
    const v8u a = *(const v8u*)(v + (2 * tokp) * 32 + d0v);
    const v8u c = *(const v8u*)(v + (2 * tokp + 1) * 32 + d0v);
    #pragma unroll
    for (int i = 0; i < 8; ++i)
      Vt32[(d0v + i) * 32 + tokp] = pk2(a[i], c[i]);
  }

  const int wave = tid >> 5, lane = tid & 31;
  const int halfl = lane >> 4, rowl = lane & 15;

  // Phase 1: S = q @ k^T (k B-fragments are contiguous 32B per lane)
  v16u aq;
  {
    const unsigned short* qp = q + (wave*16 + rowl) * 32;
    aq = *(const v16u*)(qp + halfl * 16);
  }
  #pragma unroll
  for (int j = 0; j < 4; ++j) {
    const v16u bk = *(const v16u*)(k + (j*16 + rowl) * 32 + halfl * 16);
    v8f z = {};
    v8f s = wmma_bf16(aq, bk, z);
    #pragma unroll
    for (int r = 0; r < 8; ++r)
      Sf[wave*16 + halfl*8 + r][j*16 + rowl] = s[r];
  }
  __syncthreads();

  // Phase 2: bias + shift masks + softmax -> Pb (bf16)
  if (tid < 64) {
    const int row = tid, ix = row >> 3, iy = row & 7;
    const bool ulw = (win == 56);          // dots.at[:, :, -nWW] (single index)
    const bool lrw = ((win & 7) == 7);     // dots.at[:, :, nWW-1::nWW]
    float mx = -3.0e38f;
    for (int j = 0; j < 64; ++j) {
      float s = Sf[row][j] + pe[(ix - (j >> 3) + 7) * 15 + (iy - (j & 7) + 7)];
      bool dead = (ulw && ((row >= 32) != (j >= 32))) ||
                  (lrw && ((iy >= 4) != ((j & 7) >= 4)));
      s = dead ? -3.0e38f : s;
      Sf[row][j] = s;
      mx = fmaxf(mx, s);
    }
    float sum = 0.f;
    for (int j = 0; j < 64; ++j) { float e = __expf(Sf[row][j] - mx); Sf[row][j] = e; sum += e; }
    float inv = 1.0f / sum;
    for (int j = 0; j < 64; ++j) Pb[row][j] = f2bf_u(Sf[row][j] * inv);
  }
  __syncthreads();

  // Phase 3: O = P @ V  (V B-fragments contiguous from Vt)
  v8f oacc[2] = {};
  #pragma unroll
  for (int kc = 0; kc < 2; ++kc) {
    const unsigned short* pp = &Pb[wave*16 + rowl][kc * 32];
    v16u ap = *(const v16u*)(pp + halfl * 16);
    #pragma unroll
    for (int nj = 0; nj < 2; ++nj) {
      const int col = nj*16 + rowl;              // d
      const int koff = kc*32 + halfl*16;         // tok
      const v16u bv = *(const v16u*)(&Vt[col * 64 + koff]);  // contiguous
      oacc[nj] = wmma_bf16(ap, bv, oacc[nj]);
    }
  }
  unsigned short* ob = o + base + (size_t)(wave*16 + halfl*8) * 32 + rowl;
  #pragma unroll
  for (int nj = 0; nj < 2; ++nj)
    #pragma unroll
    for (int r = 0; r < 8; ++r)
      ob[(size_t)r * 32 + nj*16] = f2bf_u(oacc[nj][r]);
}

// ---------------------------------------------------------------------------
// Kernel 3: out = attn @ w_out + b_out, rolled back (+4,+4), f32 output.
// ---------------------------------------------------------------------------
__global__ __launch_bounds__(256)
void proj_kernel(const unsigned short* __restrict__ attn,
                 const float* __restrict__ w_out,
                 const float* __restrict__ b_out,
                 float* __restrict__ out) {
  __shared__ __align__(32) unsigned short As[64*32];
  __shared__ __align__(32) unsigned short Bsr[128*32];   // [n][k]
  unsigned int* Bsr32 = (unsigned int*)Bsr;              // [n][kpair]
  const int tid = threadIdx.x;
  const int mt = blockIdx.x;
  const int n0 = blockIdx.y * 128;
  const int lane = tid & 31, wave = tid >> 5;
  const int wm = wave >> 2, wn = wave & 3;
  const int halfl = lane >> 4, rowl = lane & 15;

  v8f acc[2][2] = {};

  const int sr = tid >> 2, sc = (tid & 3) * 8;
  const int m_s = mt * 64 + sr;
  const int bb = m_s >> 12, hh = (m_s >> 6) & 63, ww = m_s & 63;
  const int win = ((hh >> 3) << 3) + (ww >> 3);
  const int tok = ((hh & 7) << 3) + (ww & 7);
  const size_t abase = (((size_t)(bb * 8) * 64 + win) * 64 + tok) * 32 + sc;
  const size_t astep = (size_t)64 * 64 * 32;   // per-head stride (head == kc)

  const int bkp = tid >> 4, bn0 = (tid & 15) * 8;
  const float* bsrc0 = w_out + (size_t)(2 * bkp) * 256 + n0 + bn0;

  for (int kc = 0; kc < 8; ++kc) {
    if (kc < 7) {
      __builtin_prefetch(attn + abase + (size_t)(kc + 1) * astep, 0, 3);
      __builtin_prefetch(bsrc0 + (size_t)(kc + 1) * 32 * 256, 0, 3);
    }
    { // stage A (already bf16, contiguous 16B per thread)
      v8u av = *(const v8u*)(attn + abase + (size_t)kc * astep);
      *(v8u*)(&As[sr * 32 + sc]) = av;
    }
    { // stage B transposed with packed K-pairs: Bsr32[n][kp]
      const float4* p0 = (const float4*)(bsrc0 + (size_t)kc * 32 * 256);
      const float4* p1 = (const float4*)(bsrc0 + (size_t)kc * 32 * 256 + 256);
      float4 a0 = p0[0], a1 = p0[1];
      float4 b0 = p1[0], b1 = p1[1];
      Bsr32[(bn0 + 0) * 16 + bkp] = pkbf2(a0.x, b0.x);
      Bsr32[(bn0 + 1) * 16 + bkp] = pkbf2(a0.y, b0.y);
      Bsr32[(bn0 + 2) * 16 + bkp] = pkbf2(a0.z, b0.z);
      Bsr32[(bn0 + 3) * 16 + bkp] = pkbf2(a0.w, b0.w);
      Bsr32[(bn0 + 4) * 16 + bkp] = pkbf2(a1.x, b1.x);
      Bsr32[(bn0 + 5) * 16 + bkp] = pkbf2(a1.y, b1.y);
      Bsr32[(bn0 + 6) * 16 + bkp] = pkbf2(a1.z, b1.z);
      Bsr32[(bn0 + 7) * 16 + bkp] = pkbf2(a1.w, b1.w);
    }
    __syncthreads();
    v16u afrag[2];
    #pragma unroll
    for (int mi = 0; mi < 2; ++mi) {
      const unsigned short* ap = &As[(wm*32 + mi*16 + rowl) * 32];
      afrag[mi] = *(const v16u*)(ap + halfl * 16);
    }
    #pragma unroll
    for (int nj = 0; nj < 2; ++nj) {
      const int bc = wn*32 + nj*16 + rowl;
      const v16u bu = *(const v16u*)(&Bsr[bc * 32 + halfl * 16]);
      #pragma unroll
      for (int mi = 0; mi < 2; ++mi)
        acc[mi][nj] = wmma_bf16(afrag[mi], bu, acc[mi][nj]);
    }
    __syncthreads();
  }
  // Epilogue with hoisted address math; rows are consecutive w within an
  // 8-aligned run, roll-back handled via (w2 + r + 4) & 63.
  #pragma unroll
  for (int mi = 0; mi < 2; ++mi) {
    const int gm0 = mt*64 + wm*32 + mi*16 + halfl*8;   // multiple of 8
    const int b2 = gm0 >> 12, h2 = (gm0 >> 6) & 63, w2 = gm0 & 63;
    const int hd = (h2 + DISP) & 63;
    const size_t rowbase = (size_t)((b2 << 12) + (hd << 6)) * 256;
    #pragma unroll
    for (int nj = 0; nj < 2; ++nj) {
      const int gn = n0 + wn*32 + nj*16 + rowl;
      const float bias = b_out[gn];
      #pragma unroll
      for (int r = 0; r < 8; ++r) {
        const int wdd = (w2 + r + DISP) & 63;          // roll-back
        out[rowbase + (size_t)wdd * 256 + gn] = acc[mi][nj][r] + bias;
      }
    }
  }
}

// ---------------------------------------------------------------------------
extern "C" void kernel_launch(void* const* d_in, const int* in_sizes, int n_in,
                              void* d_out, int out_size, void* d_ws, size_t ws_size,
                              hipStream_t stream) {
  (void)in_sizes; (void)n_in; (void)out_size; (void)ws_size;
  const float* x       = (const float*)d_in[0];
  const float* w_qkv   = (const float*)d_in[1];
  const float* pos_emb = (const float*)d_in[2];
  const float* w_out   = (const float*)d_in[3];
  const float* b_out   = (const float*)d_in[4];
  float* out = (float*)d_out;

  // workspace: q/k/v bf16 (3 x 64MB) + attention output bf16 (64MB) = 256MB
  unsigned short* qkv_ws  = (unsigned short*)d_ws;
  unsigned short* attn_ws = qkv_ws + (size_t)3 * TSZ;

  dim3 gA(2048, 6);   // M tiles x N tiles (768/128)
  qkv_kernel<<<gA, 256, 0, stream>>>(x, w_qkv, qkv_ws);

  attn_kernel<<<32*8*64, 128, 0, stream>>>(qkv_ws, pos_emb, attn_ws);

  dim3 gC(2048, 2);   // M tiles x N tiles (256/128)
  proj_kernel<<<gC, 256, 0, stream>>>(attn_ws, w_out, b_out, out);
}
